// AutoEncoder_6433861009632
// MI455X (gfx1250) — matrix operand order repaired and checked by reading, not compiled, not run on hardware
//
#include <hip/hip_runtime.h>
#include <hip/hip_bf16.h>
#include <math.h>

// ---------------------------------------------------------------------------
// PointNet++-style encoder for MI455X (gfx1250, wave32, WMMA).
//
//   transpose -> FPS (async global->LDS staging + wave32 shfl argmax) ->
//   gather -> ball query -> weight pre-pack (f16 WMMA B-fragment layout) ->
//   set-abstraction MLP (v_wmma_f32_16x16x32_f16, f32 accum) x3 -> FC head.
//
// Input pointer order assumed = depth-first insertion order of setup_inputs():
//  0: xyz (64,3,4096)
//  1..4:   sa1 layer0 (W 64x6,  b, g, bt)     5..8:   sa1 layer1 (W 64x64, ...)
//  9..12:  sa2 layer0 (W 128x67, ...)         13..16: sa2 layer1 (W 128x128, ...)
//  17..20: sa3 layer0 (W 256x131, ...)        21..24: sa3 layer1 (W 256x256, ...)
//  25: fc1_W(512,256) 26: fc1_b 27: gn1_g 28: gn1_b
//  29: fc2_W(512,512) 30: fc2_b 31: gn2_g 32: gn2_b
//  33: fc3_W(768,512) 34: fc3_b
// ---------------------------------------------------------------------------

typedef __attribute__((ext_vector_type(16))) _Float16 v16h;
typedef __attribute__((ext_vector_type(8)))  float    v8f;

#define BATCH 64
#define NPTS  4096
#define INV_NORM 0.99999750f   // 1/sqrt(1+5e-6)

// ------------------------------ transpose ----------------------------------
__global__ void transpose_kernel(const float* __restrict__ xyz, float* __restrict__ pts) {
    int t = blockIdx.x * blockDim.x + threadIdx.x;      // over B*N
    if (t >= BATCH * NPTS) return;
    int b = t / NPTS, n = t % NPTS;
    const float* src = xyz + (size_t)b * 3 * NPTS;
    float* dst = pts + (size_t)t * 3;
    dst[0] = src[0 * NPTS + n];
    dst[1] = src[1 * NPTS + n];
    dst[2] = src[2 * NPTS + n];
}

// --------------------------------- FPS -------------------------------------
// One workgroup per batch. Points staged AoS into LDS with async
// global->LDS B128 copies (ASYNCcnt path); argmax via wave32 shfl reduction
// (2 barriers/iteration instead of a 9-barrier LDS tree).
template <int N>
__global__ void fps_kernel(const float* __restrict__ pts, int npoint,
                           int* __restrict__ out_idx) {
    const int b = blockIdx.x, tid = threadIdx.x, T = blockDim.x;   // T = 256
    __shared__ __align__(16) float pxyz[N * 3];
    __shared__ float dist[N];
    __shared__ float rv[8];
    __shared__ int   ri[8];
    __shared__ int   far_s;

    // async copy of the batch's points, 16B chunks, tracked by ASYNCcnt
    {
        const char* src = (const char*)(pts + (size_t)b * N * 3);
        const int nchunks = (N * 3 * 4) / 16;
        for (int c = tid; c < nchunks; c += T) {
            unsigned int  loff  = (unsigned int)(size_t)(void*)&pxyz[c * 4];
            unsigned long gaddr = (unsigned long)(size_t)(src + (size_t)c * 16);
            asm volatile("global_load_async_to_lds_b128 %0, %1, off"
                         :: "v"(loff), "v"(gaddr) : "memory");
        }
        asm volatile("s_wait_asynccnt 0x0" ::: "memory");
    }
    for (int n = tid; n < N; n += T) dist[n] = 1e10f;
    if (tid == 0) far_s = 0;
    __syncthreads();

    const int lane = tid & 31, wave = tid >> 5, nwaves = T >> 5;
    for (int i = 0; i < npoint; ++i) {
        int far = far_s;
        if (tid == 0) out_idx[b * npoint + i] = far;
        float cx = pxyz[far * 3 + 0], cy = pxyz[far * 3 + 1], cz = pxyz[far * 3 + 2];
        float best = -1.0f; int bidx = 0;
        for (int n = tid; n < N; n += T) {
            float dx = pxyz[n * 3 + 0] - cx;
            float dy = pxyz[n * 3 + 1] - cy;
            float dz = pxyz[n * 3 + 2] - cz;
            float d = dx * dx + dy * dy + dz * dz;
            float nd = fminf(dist[n], d);
            dist[n] = nd;
            if (nd > best) { best = nd; bidx = n; }
        }
        // wave32 argmax (first-index tie-break)
        #pragma unroll
        for (int off = 16; off > 0; off >>= 1) {
            float ov = __shfl_xor(best, off, 32);
            int   oi = __shfl_xor(bidx, off, 32);
            if (ov > best || (ov == best && oi < bidx)) { best = ov; bidx = oi; }
        }
        if (lane == 0) { rv[wave] = best; ri[wave] = bidx; }
        __syncthreads();
        if (tid == 0) {
            float bb = rv[0]; int bi = ri[0];
            for (int w = 1; w < nwaves; ++w)
                if (rv[w] > bb || (rv[w] == bb && ri[w] < bi)) { bb = rv[w]; bi = ri[w]; }
            far_s = bi;
        }
        __syncthreads();
    }
}

// ------------------------------- gather ------------------------------------
__global__ void gather_xyz_kernel(const float* __restrict__ pts, const int* __restrict__ idx,
                                  int N, int S, float* __restrict__ out) {
    int t = blockIdx.x * blockDim.x + threadIdx.x;      // over B*S*3
    if (t >= BATCH * S * 3) return;
    int c = t % 3, bs = t / 3;
    int b = bs / S, s = bs % S;
    out[t] = pts[((size_t)b * N + idx[b * S + s]) * 3 + c];
}

// ----------------------------- ball query ----------------------------------
__global__ void ball_query_kernel(const float* __restrict__ pts, const float* __restrict__ centers,
                                  int N, int S, int K, float r2, int* __restrict__ idx) {
    int t = blockIdx.x * blockDim.x + threadIdx.x;      // over B*S
    if (t >= BATCH * S) return;
    int b = t / S;
    float cx = centers[t * 3 + 0], cy = centers[t * 3 + 1], cz = centers[t * 3 + 2];
    const float* P = pts + (size_t)b * N * 3;
    int cnt = 0, first = 0;
    for (int n = 0; n < N && cnt < K; ++n) {
        float dx = P[n * 3 + 0] - cx, dy = P[n * 3 + 1] - cy, dz = P[n * 3 + 2] - cz;
        float d = dx * dx + dy * dy + dz * dz;
        if (d <= r2) { if (cnt == 0) first = n; idx[(size_t)t * K + cnt] = n; ++cnt; }
    }
    for (; cnt < K; ++cnt) idx[(size_t)t * K + cnt] = first;
}

// --------------------------- weight pre-pack -------------------------------
// Convert a row-major OxCIN fp32 weight matrix into the per-lane f16 WMMA
// B-fragment layout: for tile (nt, kkidx), lane l holds 16 contiguous halves
// = W[nt*16 + (l&15)][kkidx*32 + 8*(l>>4) + (e<8 ? e : e+8)]  (zero beyond CIN).
// One aligned 32B vector load per B fragment in the MLP kernels.
__global__ void pack_w_kernel(const float* __restrict__ W, int CIN, int KT, int ntiles,
                              _Float16* __restrict__ out) {
    int t = blockIdx.x * blockDim.x + threadIdx.x;      // over ntiles*32
    if (t >= ntiles * 32) return;
    int tile = t >> 5, lane = t & 31;
    int nt = tile / KT, kkidx = tile % KT;
    const float* row = W + (size_t)(nt * 16 + (lane & 15)) * CIN;
    int cbase = kkidx * 32 + 8 * (lane >> 4);
    _Float16* dst = out + (size_t)t * 16;
#pragma unroll
    for (int e = 0; e < 16; ++e) {
        int c = cbase + ((e < 8) ? e : e + 8);
        dst[e] = (_Float16)((c < CIN) ? row[c] : 0.0f);
    }
}

// ------------------------ WMMA fragment helpers ----------------------------
// A fragment: 16-bit A 16x32 lane layout (ISA 7.12.2); two b128 LDS loads.
__device__ __forceinline__ v16h frag_a_lds(const _Float16* M, int ld, int row0, int col0) {
    int lane = threadIdx.x & 31;
    int m = lane & 15, hi = lane >> 4;
    v16h a;
#pragma unroll
    for (int e = 0; e < 16; ++e) {
        int k = (e < 8) ? (e + 8 * hi) : (e + 8 + 8 * hi);
        a[e] = M[(row0 + m) * ld + col0 + k];
    }
    return a;
}

__device__ __forceinline__ v16h frag_b_packed(const _Float16* Wp, int tile) {
    int lane = threadIdx.x & 31;
    return *(const v16h*)(Wp + ((size_t)tile * 32 + lane) * 16);
}

// ------------------------ set-abstraction MLP+max --------------------------
// One workgroup (4 wave32) per (batch, center). G and H1 staged in LDS as
// f16; both matmuls on v_wmma_f32_16x16x32_f16 with f32 accumulate.
// Layer-2 tiles max-reduce in-register (v8f + shfl_xor 16) into per-wave
// LDS stripes -> final max pooling (no H2 materialization).
template <int K, int CIN, int CINP, int C1, int C2, bool RECENTER>
__global__ void sa_mlp_kernel(const float* __restrict__ pts, int N,
                              const float* __restrict__ feats, int CF,
                              const float* __restrict__ centers,
                              const int* __restrict__ idx, int S,
                              const _Float16* __restrict__ W1p, const float* __restrict__ b1,
                              const float* __restrict__ g1, const float* __restrict__ bt1,
                              const _Float16* __restrict__ W2p, const float* __restrict__ b2,
                              const float* __restrict__ g2, const float* __restrict__ bt2,
                              float* __restrict__ out) {
    constexpr int NW  = 4;                 // 4 waves / 128 threads
    constexpr int KT1 = CINP / 32, KT2 = C1 / 32;
    constexpr int NT1 = C1 / 16,   NT2 = C2 / 16;
    const int bs = blockIdx.x;
    const int b = bs / S;
    const int tid = threadIdx.x;
    const int wave = tid >> 5;

    __shared__ _Float16 Gs[K * CINP];      // grouped + centered input
    __shared__ _Float16 H1s[K * C1];       // layer-1 activations
    __shared__ float    maxb[NW * C2];     // per-wave max stripes

    // pull packed layer-2 weights toward the caches while we stage G
    for (int p = tid * 64; p < NT2 * KT2 * 512; p += blockDim.x * 64)
        __builtin_prefetch(&W2p[p], 0, 1);

    // ---- stage grouped input (zero-pad K-dim to CINP) ----
    float ccx = 0.f, ccy = 0.f, ccz = 0.f;
    if (RECENTER) { ccx = centers[bs * 3 + 0]; ccy = centers[bs * 3 + 1]; ccz = centers[bs * 3 + 2]; }
    for (int t = tid; t < K * CINP; t += blockDim.x) {
        int k = t / CINP, c = t % CINP;
        int pid = RECENTER ? idx[(size_t)bs * K + k] : k;
        float v = 0.0f;
        if (c < 3) {
            v = pts[((size_t)b * N + pid) * 3 + c];
            if (RECENTER) v -= (c == 0 ? ccx : (c == 1 ? ccy : ccz));
        } else if (c < CIN) {
            v = feats[((size_t)b * N + pid) * CF + (c - 3)];
        }
        Gs[t] = (_Float16)v;
    }
    for (int t = tid; t < NW * C2; t += blockDim.x) maxb[t] = 0.0f;  // relu => max >= 0
    __syncthreads();

    const int lane = tid & 31;
    const int nn = lane & 15, hh = lane >> 4;

    // ---- layer 1: H1 = relu(g*((G @ W1^T + b)*inv) + bt) ----
    for (int t = wave; t < (K / 16) * NT1; t += NW) {
        int mt = t / NT1, nt = t % NT1;
        v8f acc = {};
#pragma unroll
        for (int kk = 0; kk < KT1; ++kk) {
            v16h a  = frag_a_lds(Gs, CINP, mt * 16, kk * 32);
            v16h bf = frag_b_packed(W1p, nt * KT1 + kk);
            acc = __builtin_amdgcn_wmma_f32_16x16x32_f16(false, a, false, bf,
                                                         (short)0, acc, false, false);
        }
        int o = nt * 16 + nn;
        float bb = b1[o], gg = g1[o], bt = bt1[o];
#pragma unroll
        for (int v = 0; v < 8; ++v) {
            float h = gg * ((acc[v] + bb) * INV_NORM) + bt;
            H1s[(mt * 16 + v + 8 * hh) * C1 + o] = (_Float16)fmaxf(h, 0.0f);
        }
    }
    __syncthreads();

    // ---- layer 2 + max pooling over K ----
    for (int t = wave; t < (K / 16) * NT2; t += NW) {
        int mt = t / NT2, nt = t % NT2;
        v8f acc = {};
#pragma unroll
        for (int kk = 0; kk < KT2; ++kk) {
            v16h a  = frag_a_lds(H1s, C1, mt * 16, kk * 32);
            v16h bf = frag_b_packed(W2p, nt * KT2 + kk);
            acc = __builtin_amdgcn_wmma_f32_16x16x32_f16(false, a, false, bf,
                                                         (short)0, acc, false, false);
        }
        int o = nt * 16 + nn;
        float bb = b2[o], gg = g2[o], bt = bt2[o];
        float vmax = 0.0f;
#pragma unroll
        for (int v = 0; v < 8; ++v) {
            float h = gg * ((acc[v] + bb) * INV_NORM) + bt;
            vmax = fmaxf(vmax, fmaxf(h, 0.0f));
        }
        vmax = fmaxf(vmax, __shfl_xor(vmax, 16, 32));     // merge M halves (wave32)
        if (lane < 16) {
            float* slot = &maxb[wave * C2 + o];
            *slot = fmaxf(*slot, vmax);
        }
    }
    __syncthreads();

    for (int o = tid; o < C2; o += blockDim.x) {
        float m = maxb[o];
#pragma unroll
        for (int w = 1; w < NW; ++w) m = fmaxf(m, maxb[w * C2 + o]);
        out[(size_t)bs * C2 + o] = m;
    }
}

// ------------------------------- FC head -----------------------------------
__global__ void head_kernel(const float* __restrict__ l3,
                            const float* __restrict__ W1, const float* __restrict__ b1,
                            const float* __restrict__ g1, const float* __restrict__ bt1,
                            const float* __restrict__ W2, const float* __restrict__ b2,
                            const float* __restrict__ g2, const float* __restrict__ bt2,
                            const float* __restrict__ W3, const float* __restrict__ b3,
                            float* __restrict__ out) {
    const int b = blockIdx.x, tid = threadIdx.x;       // 512 threads
    __shared__ float x[512];
    __shared__ float rsum[512], rsq[512];

    if (tid < 256) x[tid] = l3[b * 256 + tid];
    __syncthreads();

    float acc = b1[tid];
    for (int c = 0; c < 256; ++c) acc += x[c] * W1[tid * 256 + c];
    rsum[tid] = acc; rsq[tid] = acc * acc;
    __syncthreads();
    for (int s = 256; s > 0; s >>= 1) {
        if (tid < s) { rsum[tid] += rsum[tid + s]; rsq[tid] += rsq[tid + s]; }
        __syncthreads();
    }
    {
        float mean = rsum[0] / 512.0f;
        float var  = rsq[0] / 512.0f - mean * mean;
        float h = (acc - mean) * rsqrtf(var + 1e-5f) * g1[tid] + bt1[tid];
        __syncthreads();
        x[tid] = fmaxf(h, 0.0f);
    }
    __syncthreads();

    acc = b2[tid];
    for (int c = 0; c < 512; ++c) acc += x[c] * W2[tid * 512 + c];
    rsum[tid] = acc; rsq[tid] = acc * acc;
    __syncthreads();
    for (int s = 256; s > 0; s >>= 1) {
        if (tid < s) { rsum[tid] += rsum[tid + s]; rsq[tid] += rsq[tid + s]; }
        __syncthreads();
    }
    {
        float mean = rsum[0] / 512.0f;
        float var  = rsq[0] / 512.0f - mean * mean;
        float h = (acc - mean) * rsqrtf(var + 1e-5f) * g2[tid] + bt2[tid];
        __syncthreads();
        x[tid] = fmaxf(h, 0.0f);
    }
    __syncthreads();

    for (int o = tid; o < 768; o += 512) {
        float a = b3[o];
        for (int c = 0; c < 512; ++c) a += x[c] * W3[o * 512 + c];
        out[(size_t)b * 768 + o] = a;
    }
}

// ------------------------------- launcher ----------------------------------
extern "C" void kernel_launch(void* const* d_in, const int* in_sizes, int n_in,
                              void* d_out, int out_size, void* d_ws, size_t ws_size,
                              hipStream_t stream) {
    (void)in_sizes; (void)n_in; (void)out_size; (void)ws_size;
    const float* xyz = (const float*)d_in[0];
    const float *sa1W1 = (const float*)d_in[1],  *sa1b1 = (const float*)d_in[2],
                *sa1g1 = (const float*)d_in[3],  *sa1t1 = (const float*)d_in[4],
                *sa1W2 = (const float*)d_in[5],  *sa1b2 = (const float*)d_in[6],
                *sa1g2 = (const float*)d_in[7],  *sa1t2 = (const float*)d_in[8];
    const float *sa2W1 = (const float*)d_in[9],  *sa2b1 = (const float*)d_in[10],
                *sa2g1 = (const float*)d_in[11], *sa2t1 = (const float*)d_in[12],
                *sa2W2 = (const float*)d_in[13], *sa2b2 = (const float*)d_in[14],
                *sa2g2 = (const float*)d_in[15], *sa2t2 = (const float*)d_in[16];
    const float *sa3W1 = (const float*)d_in[17], *sa3b1 = (const float*)d_in[18],
                *sa3g1 = (const float*)d_in[19], *sa3t1 = (const float*)d_in[20],
                *sa3W2 = (const float*)d_in[21], *sa3b2 = (const float*)d_in[22],
                *sa3g2 = (const float*)d_in[23], *sa3t2 = (const float*)d_in[24];
    const float *fc1W = (const float*)d_in[25], *fc1b = (const float*)d_in[26],
                *gn1g = (const float*)d_in[27], *gn1b = (const float*)d_in[28],
                *fc2W = (const float*)d_in[29], *fc2b = (const float*)d_in[30],
                *gn2g = (const float*)d_in[31], *gn2b = (const float*)d_in[32],
                *fc3W = (const float*)d_in[33], *fc3b = (const float*)d_in[34];
    float* out = (float*)d_out;

    char* ws = (char*)d_ws;
    size_t off = 0;
    auto carve = [&](size_t bytes) -> char* {
        char* p = ws + off;
        off = (off + bytes + 255) & ~(size_t)255;
        return p;
    };
    float* pts      = (float*)carve((size_t)BATCH * NPTS * 3 * 4);
    int*   fps1_idx = (int*)  carve((size_t)BATCH * 256 * 4);
    float* l1_xyz   = (float*)carve((size_t)BATCH * 256 * 3 * 4);
    int*   ball1    = (int*)  carve((size_t)BATCH * 256 * 32 * 4);
    float* l1_p     = (float*)carve((size_t)BATCH * 256 * 64 * 4);
    int*   fps2_idx = (int*)  carve((size_t)BATCH * 128 * 4);
    float* l2_xyz   = (float*)carve((size_t)BATCH * 128 * 3 * 4);
    int*   ball2    = (int*)  carve((size_t)BATCH * 128 * 64 * 4);
    float* l2_p     = (float*)carve((size_t)BATCH * 128 * 128 * 4);
    float* l3       = (float*)carve((size_t)BATCH * 256 * 4);
    // packed f16 weights: tiles = (O/16) * (CINP/32); 512 halves per tile
    _Float16* p1a = (_Float16*)carve((size_t)4   * 1 * 512 * 2);   // 64x6   -> CINP 32
    _Float16* p1b = (_Float16*)carve((size_t)4   * 2 * 512 * 2);   // 64x64
    _Float16* p2a = (_Float16*)carve((size_t)8   * 3 * 512 * 2);   // 128x67 -> CINP 96
    _Float16* p2b = (_Float16*)carve((size_t)8   * 4 * 512 * 2);   // 128x128
    _Float16* p3a = (_Float16*)carve((size_t)16  * 5 * 512 * 2);   // 256x131-> CINP 160
    _Float16* p3b = (_Float16*)carve((size_t)16  * 8 * 512 * 2);   // 256x256

    // 0) pre-pack all SA weights into WMMA B-fragment layout
    pack_w_kernel<<<(4 * 1 * 32 + 255) / 256, 256, 0, stream>>>(sa1W1, 6,   1, 4 * 1,  p1a);
    pack_w_kernel<<<(4 * 2 * 32 + 255) / 256, 256, 0, stream>>>(sa1W2, 64,  2, 4 * 2,  p1b);
    pack_w_kernel<<<(8 * 3 * 32 + 255) / 256, 256, 0, stream>>>(sa2W1, 67,  3, 8 * 3,  p2a);
    pack_w_kernel<<<(8 * 4 * 32 + 255) / 256, 256, 0, stream>>>(sa2W2, 128, 4, 8 * 4,  p2b);
    pack_w_kernel<<<(16 * 5 * 32 + 255) / 256, 256, 0, stream>>>(sa3W1, 131, 5, 16 * 5, p3a);
    pack_w_kernel<<<(16 * 8 * 32 + 255) / 256, 256, 0, stream>>>(sa3W2, 256, 8, 16 * 8, p3b);

    // 1) transpose (B,3,N) -> (B,N,3)
    transpose_kernel<<<(BATCH * NPTS + 255) / 256, 256, 0, stream>>>(xyz, pts);

    // 2) SA1: FPS 4096->256, gather, ball(r=0.2,K=32), MLP 6->64->64 + max
    fps_kernel<NPTS><<<BATCH, 256, 0, stream>>>(pts, 256, fps1_idx);
    gather_xyz_kernel<<<(BATCH * 256 * 3 + 255) / 256, 256, 0, stream>>>(pts, fps1_idx, NPTS, 256, l1_xyz);
    ball_query_kernel<<<(BATCH * 256 + 255) / 256, 256, 0, stream>>>(pts, l1_xyz, NPTS, 256, 32, 0.04f, ball1);
    sa_mlp_kernel<32, 6, 32, 64, 64, true><<<BATCH * 256, 128, 0, stream>>>(
        pts, NPTS, pts, 3, l1_xyz, ball1, 256,
        p1a, sa1b1, sa1g1, sa1t1, p1b, sa1b2, sa1g2, sa1t2, l1_p);

    // 3) SA2: FPS 256->128, gather, ball(r=0.4,K=64), MLP 67->128->128 + max
    fps_kernel<256><<<BATCH, 256, 0, stream>>>(l1_xyz, 128, fps2_idx);
    gather_xyz_kernel<<<(BATCH * 128 * 3 + 255) / 256, 256, 0, stream>>>(l1_xyz, fps2_idx, 256, 128, l2_xyz);
    ball_query_kernel<<<(BATCH * 128 + 255) / 256, 256, 0, stream>>>(l1_xyz, l2_xyz, 256, 128, 64, 0.16f, ball2);
    sa_mlp_kernel<64, 67, 96, 128, 128, true><<<BATCH * 128, 128, 0, stream>>>(
        l1_xyz, 256, l1_p, 64, l2_xyz, ball2, 128,
        p2a, sa2b1, sa2g1, sa2t1, p2b, sa2b2, sa2g2, sa2t2, l2_p);

    // 4) SA3: group-all, MLP 131->256->256 + max over 128 points
    sa_mlp_kernel<128, 131, 160, 256, 256, false><<<BATCH, 128, 0, stream>>>(
        l2_xyz, 128, l2_p, 128, nullptr, nullptr, 1,
        p3a, sa3b1, sa3g1, sa3t1, p3b, sa3b2, sa3g2, sa3t2, l3);

    // 5) FC head
    head_kernel<<<BATCH, 512, 0, stream>>>(l3, fc1W, fc1b, gn1g, gn1b,
                                           fc2W, fc2b, gn2g, gn2b, fc3W, fc3b, out);
}
